// GAT_37495064494702
// MI455X (gfx1250) — matrix operand order repaired and checked by reading, not compiled, not run on hardware
//
#include <hip/hip_runtime.h>
#include <hip/hip_bf16.h>

#define HEADS  4
#define IN_DIM 256
#define HID    64
#define NCLS   40

typedef __attribute__((ext_vector_type(16))) __bf16 bf16x16;
typedef __attribute__((ext_vector_type(8)))  __bf16 bf16x8;
typedef __attribute__((ext_vector_type(8)))  float  f32x8;

// ---------------- helpers ----------------
__device__ __forceinline__ __bf16 f2bf(float f) {
    unsigned u = __float_as_uint(f);
    u += 0x7FFFu + ((u >> 16) & 1u);          // round-to-nearest-even
    unsigned short h = (unsigned short)(u >> 16);
    __bf16 r;
    __builtin_memcpy(&r, &h, 2);
    return r;
}
__device__ __forceinline__ float lrelu(float x) { return x > 0.f ? x : 0.01f * x; }
__device__ __forceinline__ float eluf(float x)  { return x > 0.f ? x : __expf(x) - 1.f; }

// monotone order-preserving float<->uint encoding (for atomic max on floats)
__device__ __forceinline__ unsigned enc_f(float f) {
    unsigned u = __float_as_uint(f);
    return (u & 0x80000000u) ? ~u : (u | 0x80000000u);
}
__device__ __forceinline__ float dec_f(unsigned u) {
    return (u & 0x80000000u) ? __uint_as_float(u & 0x7FFFFFFFu) : __uint_as_float(~u);
}

// ---------------- conversion / packing ----------------
__global__ void cvt_bf16_kernel(const float* __restrict__ in, __bf16* __restrict__ out, long long n) {
    long long t = (long long)blockIdx.x * blockDim.x + threadIdx.x;
    if (t < n) out[t] = f2bf(in[t]);
}

// W: [HEADS, din, dout] f32  ->  Wt: [HEADS*dout, din] bf16 (column-major B for WMMA)
__global__ void pack_w_kernel(const float* __restrict__ W, __bf16* __restrict__ Wt, int din, int dout) {
    int t = blockIdx.x * blockDim.x + threadIdx.x;
    int tot = HEADS * din * dout;
    if (t >= tot) return;
    int h   = t / (din * dout);
    int rem = t - h * (din * dout);
    int k   = rem / dout;
    int j   = rem - k * dout;
    Wt[(size_t)(h * dout + j) * din + k] = f2bf(W[t]);
}

// ---------------- WMMA GEMM: C[M,Ncols] = A[M,K](bf16) x Bt[Ncols,K](bf16) (+bias) (+Cin) ----------------
__global__ __launch_bounds__(256)
void wmma_gemm_kernel(const __bf16* __restrict__ A, const __bf16* __restrict__ Bt,
                      const float* __restrict__ bias, const float* __restrict__ Cin,
                      float* __restrict__ Cout, int M, int Ncols, int K) {
    int wid  = threadIdx.x >> 5;
    int lane = threadIdx.x & 31;
    int m0 = blockIdx.x * 32 + (wid >> 2) * 16;
    int n0 = blockIdx.y * 64 + (wid & 3) * 16;
    if (m0 >= M || n0 >= Ncols) return;        // wave-uniform exit (EXEC stays all-ones for WMMA)

    int half = lane >> 4;                      // 0: lanes 0-15, 1: lanes 16-31
    int l16  = lane & 15;
    int arow = m0 + l16;                       // A: lane = M row
    int col  = n0 + l16;                       // B/C/D: lane = N column

    const __bf16* Aptr = A  + (size_t)arow * K + half * 16;   // K rows 0..15 / 16..31, contiguous
    const __bf16* Bptr = Bt + (size_t)col  * K + half * 16;   // K rows 0..15 / 16..31, contiguous

    f32x8 acc = {};
    if (Cin) {
        const float* cp = Cin + (size_t)(m0 + half * 8) * Ncols + col;
        #pragma unroll
        for (int r = 0; r < 8; ++r) acc[r] = cp[(size_t)r * Ncols];
    }

    #pragma unroll
    for (int k0 = 0; k0 < 256; k0 += 32) {
        bf16x8 alo = *(const bf16x8*)(Aptr + k0);        // K = base+0..7
        bf16x8 ahi = *(const bf16x8*)(Aptr + k0 + 8);    // K = base+8..15
        bf16x16 af = __builtin_shufflevector(alo, ahi, 0,1,2,3,4,5,6,7,8,9,10,11,12,13,14,15);
        bf16x16 bf = *(const bf16x16*)(Bptr + k0);
        acc = __builtin_amdgcn_wmma_f32_16x16x32_bf16(false, af, false, bf, (short)0, acc, false, false);
    }

    float bv = bias ? bias[col] : 0.0f;
    float* op = Cout + (size_t)(m0 + half * 8) * Ncols + col;
    #pragma unroll
    for (int r = 0; r < 8; ++r) op[(size_t)r * Ncols] = acc[r] + bv;
}

// ---------------- attention scores: a1/a2[n,h] = ft_h[n,:] . al/ar[h] + bias ----------------
__global__ void ascore_kernel(const float* __restrict__ ft,
                              const float* __restrict__ al, const float* __restrict__ alb,
                              const float* __restrict__ ar, const float* __restrict__ arb,
                              float* __restrict__ a1, float* __restrict__ a2,
                              int N, int dout, int HD) {
    int n    = blockIdx.x * (blockDim.x >> 5) + (threadIdx.x >> 5);
    int lane = threadIdx.x & 31;
    if (n >= N) return;
    const float* row = ft + (size_t)n * HD;
    for (int h = 0; h < HEADS; ++h) {
        float s1 = 0.f, s2 = 0.f;
        const float* fh  = row + h * dout;
        const float* alh = al  + h * dout;
        const float* arh = ar  + h * dout;
        for (int j = lane; j < dout; j += 32) {
            float v = fh[j];
            s1 += v * alh[j];
            s2 += v * arh[j];
        }
        #pragma unroll
        for (int m = 16; m; m >>= 1) { s1 += __shfl_xor(s1, m, 32); s2 += __shfl_xor(s2, m, 32); }
        if (lane == 0) {
            a1[n * HEADS + h] = s1 + alb[h];
            a2[n * HEADS + h] = s2 + arb[h];
        }
    }
}

// ---------------- edge pass 1: per-destination segment max ----------------
__global__ void edge_max_kernel(const int* __restrict__ src, const int* __restrict__ dst,
                                const float4* __restrict__ a1, const float4* __restrict__ a2,
                                unsigned* __restrict__ emax, int E) {
    int i = blockIdx.x * blockDim.x + threadIdx.x;
    if (i >= E) return;
    int s = src[i], d = dst[i];
    float4 x = a1[d], y = a2[s];
    atomicMax(&emax[d * 4 + 0], enc_f(lrelu(x.x + y.x)));
    atomicMax(&emax[d * 4 + 1], enc_f(lrelu(x.y + y.y)));
    atomicMax(&emax[d * 4 + 2], enc_f(lrelu(x.z + y.z)));
    atomicMax(&emax[d * 4 + 3], enc_f(lrelu(x.w + y.w)));
}

// ---------------- edge pass 2: ee = exp(e - max), denom += ee ----------------
__global__ void edge_exp_kernel(const int* __restrict__ src, const int* __restrict__ dst,
                                const float4* __restrict__ a1, const float4* __restrict__ a2,
                                const unsigned* __restrict__ emax, float* __restrict__ denom,
                                float4* __restrict__ ee, int E) {
    int i = blockIdx.x * blockDim.x + threadIdx.x;
    if (i >= E) return;
    int s = src[i], d = dst[i];
    float4 x = a1[d], y = a2[s];
    float4 r;
    r.x = __expf(lrelu(x.x + y.x) - dec_f(emax[d * 4 + 0]));
    r.y = __expf(lrelu(x.y + y.y) - dec_f(emax[d * 4 + 1]));
    r.z = __expf(lrelu(x.z + y.z) - dec_f(emax[d * 4 + 2]));
    r.w = __expf(lrelu(x.w + y.w) - dec_f(emax[d * 4 + 3]));
    ee[i] = r;
    atomicAdd(&denom[d * 4 + 0], r.x);
    atomicAdd(&denom[d * 4 + 1], r.y);
    atomicAdd(&denom[d * 4 + 2], r.z);
    atomicAdd(&denom[d * 4 + 3], r.w);
}

// ---------------- edge pass 3: accum[dst,c] += alpha * ft[src,c] ----------------
__global__ void edge_accum_kernel(const int* __restrict__ src, const int* __restrict__ dst,
                                  const float* __restrict__ ee, const float* __restrict__ denom,
                                  const float* __restrict__ ft, float* __restrict__ accum,
                                  int E, int HD, int dout) {
    long long t = (long long)blockIdx.x * blockDim.x + threadIdx.x;
    long long tot = (long long)E * HD;
    if (t >= tot) return;
    int i = (int)(t / HD);
    int c = (int)(t - (long long)i * HD);
    int h = c / dout;
    int s = src[i], d = dst[i];
    float alpha = ee[i * 4 + h] / denom[d * 4 + h];
    atomicAdd(&accum[(size_t)d * HD + c], alpha * ft[(size_t)s * HD + c]);
}

// ---------------- elu + bf16 convert (next-layer input) ----------------
__global__ void elu_cvt_kernel(const float* __restrict__ in, __bf16* __restrict__ out, long long n) {
    long long t = (long long)blockIdx.x * blockDim.x + threadIdx.x;
    if (t < n) out[t] = f2bf(eluf(in[t]));
}

// ---------------- final: sum-of-heads elu, gather train nodes ----------------
__global__ void final_kernel(const float* __restrict__ accum, const int* __restrict__ train,
                             float* __restrict__ out, int NT) {
    int t = blockIdx.x * blockDim.x + threadIdx.x;
    if (t >= NT * NCLS) return;
    int ti = t / NCLS, j = t - ti * NCLS;
    int n = train[ti];
    float s = 0.f;
    #pragma unroll
    for (int h = 0; h < HEADS; ++h) s += eluf(accum[(size_t)n * (HEADS * NCLS) + h * NCLS + j]);
    out[t] = s;
}

// ---------------- driver ----------------
extern "C" void kernel_launch(void* const* d_in, const int* in_sizes, int n_in,
                              void* d_out, int out_size, void* d_ws, size_t ws_size,
                              hipStream_t stream) {
    const float* feats = (const float*)d_in[0];
    const int*   src   = (const int*)d_in[1];
    const int*   dst   = (const int*)d_in[2];
    const int*   train = (const int*)d_in[3];
    int N  = in_sizes[0] / IN_DIM;
    int E  = in_sizes[1];
    int NT = in_sizes[3];

    const float* W[3]   = {(const float*)d_in[4],  (const float*)d_in[10], (const float*)d_in[18]};
    const float* b[3]   = {(const float*)d_in[5],  (const float*)d_in[11], (const float*)d_in[19]};
    const float* al[3]  = {(const float*)d_in[6],  (const float*)d_in[12], (const float*)d_in[20]};
    const float* alb[3] = {(const float*)d_in[7],  (const float*)d_in[13], (const float*)d_in[21]};
    const float* ar[3]  = {(const float*)d_in[8],  (const float*)d_in[14], (const float*)d_in[22]};
    const float* arb[3] = {(const float*)d_in[9],  (const float*)d_in[15], (const float*)d_in[23]};
    const float* Wr[3]  = {nullptr, (const float*)d_in[16], (const float*)d_in[24]};
    const float* br[3]  = {nullptr, (const float*)d_in[17], (const float*)d_in[25]};

    // workspace carving (256-byte aligned slabs)
    char* ws = (char*)d_ws;
    size_t off = 0;
    auto alloc = [&](size_t bytes) { size_t o = off; off += (bytes + 255) & ~(size_t)255; return o; };
    __bf16*   xbf   = (__bf16*)  (ws + alloc((size_t)N * IN_DIM * 2));
    float*    ft    = (float*)   (ws + alloc((size_t)N * 256 * 4));
    float*    accum = (float*)   (ws + alloc((size_t)N * 256 * 4));
    float*    a1    = (float*)   (ws + alloc((size_t)N * HEADS * 4));
    float*    a2    = (float*)   (ws + alloc((size_t)N * HEADS * 4));
    unsigned* emax  = (unsigned*)(ws + alloc((size_t)N * HEADS * 4));
    float*    denom = (float*)   (ws + alloc((size_t)N * HEADS * 4));
    float4*   ee    = (float4*)  (ws + alloc((size_t)E * HEADS * 4));
    __bf16* Wp[3], *Wrp[3];
    Wp[0]  = (__bf16*)(ws + alloc((size_t)HEADS * IN_DIM * HID  * 2));
    Wp[1]  = (__bf16*)(ws + alloc((size_t)HEADS * IN_DIM * HID  * 2));
    Wp[2]  = (__bf16*)(ws + alloc((size_t)HEADS * IN_DIM * NCLS * 2));
    Wrp[0] = nullptr;
    Wrp[1] = (__bf16*)(ws + alloc((size_t)HEADS * IN_DIM * HID  * 2));
    Wrp[2] = (__bf16*)(ws + alloc((size_t)HEADS * IN_DIM * NCLS * 2));

    // pack weights to bf16 column-major, convert features to bf16
    for (int l = 0; l < 3; ++l) {
        int dout = (l == 2) ? NCLS : HID;
        int tot = HEADS * IN_DIM * dout;
        pack_w_kernel<<<(tot + 255) / 256, 256, 0, stream>>>(W[l], Wp[l], IN_DIM, dout);
        if (l > 0)
            pack_w_kernel<<<(tot + 255) / 256, 256, 0, stream>>>(Wr[l], Wrp[l], IN_DIM, dout);
    }
    {
        long long n = (long long)N * IN_DIM;
        cvt_bf16_kernel<<<(int)((n + 255) / 256), 256, 0, stream>>>(feats, xbf, n);
    }

    for (int l = 0; l < 3; ++l) {
        int dout = (l == 2) ? NCLS : HID;
        int HD   = HEADS * dout;                    // 256, 256, 160

        // ft = x @ W + b   (all heads fused along columns)
        dim3 gg((N + 31) / 32, (HD + 63) / 64);
        wmma_gemm_kernel<<<gg, 256, 0, stream>>>(xbf, Wp[l], b[l], nullptr, ft, N, HD, IN_DIM);

        // attention scores
        ascore_kernel<<<(N + 7) / 8, 256, 0, stream>>>(ft, al[l], alb[l], ar[l], arb[l], a1, a2, N, dout, HD);

        // segment softmax + aggregation
        hipMemsetAsync(emax,  0, (size_t)N * HEADS * 4, stream);
        hipMemsetAsync(denom, 0, (size_t)N * HEADS * 4, stream);
        hipMemsetAsync(accum, 0, (size_t)N * HD * 4, stream);
        edge_max_kernel<<<(E + 255) / 256, 256, 0, stream>>>(src, dst, (const float4*)a1, (const float4*)a2, emax, E);
        edge_exp_kernel<<<(E + 255) / 256, 256, 0, stream>>>(src, dst, (const float4*)a1, (const float4*)a2, emax, denom, ee, E);
        long long tot = (long long)E * HD;
        edge_accum_kernel<<<(int)((tot + 255) / 256), 256, 0, stream>>>(src, dst, (const float*)ee, denom, ft, accum, E, HD, dout);

        // residual: accum += x @ Wr + br
        if (l > 0)
            wmma_gemm_kernel<<<gg, 256, 0, stream>>>(xbf, Wrp[l], br[l], accum, accum, N, HD, IN_DIM);

        if (l < 2) {
            long long n = (long long)N * HD;
            elu_cvt_kernel<<<(int)((n + 255) / 256), 256, 0, stream>>>(accum, xbf, n);
        } else {
            final_kernel<<<(NT * NCLS + 255) / 256, 256, 0, stream>>>(accum, train, (float*)d_out, NT);
        }
    }
    (void)n_in; (void)out_size; (void)ws_size;
}
